// GraphVAE_73409581023729
// MI455X (gfx1250) — matrix operand order repaired and checked by reading, not compiled, not run on hardware
//
#include <hip/hip_runtime.h>
#include <hip/hip_bf16.h>

// ---------------------------------------------------------------------------
// GraphVAE forward for MI455X (gfx1250, wave32).
//  - Encoder GEMMs: v_wmma_f32_16x16x32_f16 (f16 mul, f32 acc).
//  - MPM edge term: v_wmma_f32_16x16x4_f32 (full f32) on LDS-resident
//    80x80 tiles; 50-iteration fixed point in ONE workgroup (~132 KB of the
//    320 KB WGP LDS), deterministic tree reductions, A/B built in-prologue.
// ---------------------------------------------------------------------------

typedef __attribute__((ext_vector_type(16))) _Float16 v16h;
typedef __attribute__((ext_vector_type(8)))  float    v8f;
typedef __attribute__((ext_vector_type(2)))  float    v2f;

#define NN   80      // nodes
#define FF   64      // in features
#define HH   256     // hidden
#define ZZ   128     // latent
#define EE   800     // edges (before self loops)
#define OFF  3160    // N*(N-1)/2
#define ITER 50

// ---------------------------------------------------------------------------
// Degree / symmetric norm: deg = segment_sum(ones, dst) over 800 edges + self
// ---------------------------------------------------------------------------
__global__ void k_deg(const int* __restrict__ ei, float* __restrict__ dinv) {
    int n = threadIdx.x;
    if (n < NN) {
        int c = 1;                              // self loop
        const int* dst = ei + EE;
        for (int e = 0; e < EE; ++e) c += (dst[e] == n);
        dinv[n] = rsqrtf(fmaxf((float)c, 1e-12f));
    }
}

// ---------------------------------------------------------------------------
// WMMA GEMM: C[80,N] = A[80,K] @ B[K,N], f32 in/out, f16 multiply, f32 acc.
// One wave per 16x16 tile; 80 tiles total (20 blocks x 128 threads).
// ---------------------------------------------------------------------------
__global__ void k_gemm_wmma(const float* __restrict__ A,
                            const float* __restrict__ B,
                            float* __restrict__ C, int K, int N) {
    int wave   = (blockIdx.x * blockDim.x + threadIdx.x) >> 5;
    int lane   = threadIdx.x & 31;
    int tilesN = N >> 4;
    int tm = wave / tilesN;
    int tn = wave % tilesN;
    int half = lane >> 4;      // 0: lanes 0-15, 1: lanes 16-31
    int lidx = lane & 15;
    int m0 = tm * 16, n0 = tn * 16;

    v8f acc = {};
    for (int kb = 0; kb < K; kb += 32) {
        v16h av, bv;
        // A fragment (16x32 f16): lane -> row M, VGPR halves hold K pairs.
        int arow = (m0 + lidx) * K;
        int ka   = kb + half * 8;
#pragma unroll
        for (int e = 0; e < 8; ++e) {
            av[e]     = (_Float16)A[arow + ka + e];
            av[e + 8] = (_Float16)A[arow + ka + 16 + e];
        }
        // B fragment (32x16 f16): lane -> col N, same K order as A (8h+e, 16+8h+e).
        int kB  = kb + half * 8;
        int col = n0 + lidx;
#pragma unroll
        for (int e = 0; e < 16; ++e)
            bv[e] = (_Float16)B[(kB + e + (e >= 8 ? 8 : 0)) * N + col];

        acc = __builtin_amdgcn_wmma_f32_16x16x32_f16(
            /*neg_a=*/false, av, /*neg_b=*/false, bv,
            /*c_mod=*/(short)0, acc, /*reuse_a=*/false, /*reuse_b=*/false);
    }
    // C/D layout: VGPR r -> M = r + half*8, N = lidx.
#pragma unroll
    for (int r = 0; r < 8; ++r)
        C[(m0 + half * 8 + r) * N + n0 + lidx] = acc[r];
}

// ---------------------------------------------------------------------------
// GCN aggregation: out[n,f] = sum_{e: dst==n} H[src]*dinv[src]*dinv[n]
//                            + H[n]*dinv[n]^2 + bias[f]
// One block (256 feats) per node; edge list staged in LDS. Deterministic.
// ---------------------------------------------------------------------------
__global__ void k_agg(const float* __restrict__ Hin, const float* __restrict__ dinv,
                      const int* __restrict__ ei, const float* __restrict__ bias,
                      float* __restrict__ Hout) {
    __shared__ int   sSrc[EE];
    __shared__ int   sDst[EE];
    __shared__ float sNrm[EE];
    int n = blockIdx.x, f = threadIdx.x;
    for (int e = f; e < EE; e += blockDim.x) {
        int s = ei[e], d = ei[EE + e];
        sSrc[e] = s; sDst[e] = d;
        sNrm[e] = dinv[s] * dinv[d];
    }
    __syncthreads();
    float di = dinv[n];
    float acc = Hin[n * HH + f] * di * di;     // self loop
    for (int e = 0; e < EE; ++e)
        if (sDst[e] == n) acc += Hin[sSrc[e] * HH + f] * sNrm[e];
    Hout[n * HH + f] = acc + bias[f];
}

// ---------------------------------------------------------------------------
// BatchNorm1d (training stats, eps=1e-5) + ReLU, in place; 1 thread / column.
// ---------------------------------------------------------------------------
__global__ void k_bn_relu(float* __restrict__ H, const float* __restrict__ g,
                          const float* __restrict__ bt) {
    int f = threadIdx.x;
    float m = 0.f;
    for (int r = 0; r < NN; ++r) m += H[r * HH + f];
    m *= (1.0f / NN);
    float v = 0.f;
    for (int r = 0; r < NN; ++r) { float d = H[r * HH + f] - m; v += d * d; }
    v *= (1.0f / NN);
    float inv = rsqrtf(v + 1e-5f);
    float gg = g[f], bb = bt[f];
    for (int r = 0; r < NN; ++r) {
        float val = gg * (H[r * HH + f] - m) * inv + bb;
        H[r * HH + f] = fmaxf(val, 0.0f);
    }
}

// ---------------------------------------------------------------------------
// Fused latent + decoder hidden (single block, 256 threads):
//   gvec = mean_n Henc; mu/logvar heads; z = mu + eps*exp(lv/2) (in LDS);
//   he   = relu(z @ We1 + be1)
// ---------------------------------------------------------------------------
__global__ void k_latent_he(const float* __restrict__ Henc,
                            const float* __restrict__ Wmu, const float* __restrict__ bmu,
                            const float* __restrict__ Wlv, const float* __restrict__ blv,
                            const float* __restrict__ eps,
                            const float* __restrict__ We1, const float* __restrict__ be1,
                            float* __restrict__ he) {
    __shared__ float gv[HH];
    __shared__ float sz[ZZ];
    int t = threadIdx.x;                       // 256
    float s = 0.f;
    for (int r = 0; r < NN; ++r) s += Henc[r * HH + t];
    gv[t] = s * (1.0f / NN);
    __syncthreads();
    if (t < ZZ) {
        float mu = bmu[t], lv = blv[t];
        for (int k = 0; k < HH; ++k) {
            float g = gv[k];
            mu += g * Wmu[k * ZZ + t];
            lv += g * Wlv[k * ZZ + t];
        }
        lv = fminf(fmaxf(lv, -4.0f), 4.0f);
        sz[t] = mu + eps[t] * expf(0.5f * lv);
    }
    __syncthreads();
    float a = be1[t];
    for (int k = 0; k < ZZ; ++k) a += sz[k] * We1[k * HH + t];
    he[t] = fmaxf(a, 0.0f);
}

// vec_logits = he @ We2 + be2   [3160]
__global__ void k_vl(const float* __restrict__ he, const float* __restrict__ We2,
                     const float* __restrict__ be2, float* __restrict__ vl) {
    __shared__ float sh[HH];
    int t = threadIdx.x;
    sh[t] = he[t];
    __syncthreads();
    int o = blockIdx.x * blockDim.x + t;
    if (o < OFF) {
        float a = be2[o];
        for (int k = 0; k < HH; ++k) a += sh[k] * We2[k * OFF + o];
        vl[o] = a;
    }
}

// ---------------------------------------------------------------------------
// MPM fixed point: 50 iterations, single workgroup, all state in LDS.
// Prologue builds A = max(adj,adj^T) (diag 1) and B = sigmoid(A_hat) (diag 1)
// directly in LDS, computes degrees, then zeroes BOTH diagonals:
//   - sA diag 0  <=>  sum over j != i
//   - sB diag 0  <=>  the (b==a -> 0) mask of S, branch-free max loop
// Iteration:
//   M[j,a] = max_b sB[a,b]*X[j,b]                  (VALU mul+max stream)
//   Xn     = X.*nsim + sA @ M                      (f32 WMMA, 5x5 tiles)
//   X      = Xn / ||Xn||_F                         (deterministic LDS tree)
// ---------------------------------------------------------------------------
__global__ void k_mpm(const int* __restrict__ adj, const float* __restrict__ vl,
                      float* __restrict__ out) {
    __shared__ __align__(16) float sA[NN * NN];
    __shared__ __align__(16) float sB[NN * NN];
    __shared__ __align__(16) float sX[NN * NN];
    __shared__ __align__(16) float sM[NN * NN];
    __shared__ __align__(16) float sXn[NN * NN];
    __shared__ float sdA[NN], sdB[NN], red[1024];
    int tid = threadIdx.x, nt = blockDim.x;
    int wave = tid >> 5;
    int lane = tid & 31;
    int half = lane >> 4;
    int lidx = lane & 15;

    // Build A and B in LDS (replaces separate k_build_AB kernel).
    for (int idx = tid; idx < NN * NN; idx += nt) {
        int i = idx / NN, j = idx % NN;
        float a, b;
        if (i == j) {
            a = 1.0f; b = 1.0f;
        } else {
            a = (adj[i * NN + j] | adj[j * NN + i]) ? 1.0f : 0.0f;
            int lo = i < j ? i : j, hi = i < j ? j : i;
            int t = lo * (2 * NN - lo - 1) / 2 + (hi - lo - 1);   // triu k=1
            b = 1.0f / (1.0f + expf(-vl[t]));
        }
        sA[idx] = a;
        sB[idx] = b;
        sX[idx] = 1.0f / NN;
    }
    __syncthreads();
    if (tid < NN) {                       // degA (includes diag=1)
        float s = 0.f;
        for (int j = 0; j < NN; ++j) s += sA[tid * NN + j];
        sdA[tid] = s;
    } else if (tid < 2 * NN) {            // degB (includes diag=1)
        int a = tid - NN; float s = 0.f;
        for (int b = 0; b < NN; ++b) s += sB[a * NN + b];
        sdB[a] = s;
    }
    __syncthreads();
    if (tid < NN) sA[tid * NN + tid] = 0.0f;                 // j != i
    else if (tid < 2 * NN) { int a = tid - NN; sB[a * NN + a] = 0.0f; } // b==a -> 0
    __syncthreads();

    // WMMA tile assignment: waves 0..24 own 16x16 output tiles of the 80x80.
    int tm = wave / 5, tn = wave % 5;
    int m0 = tm * 16, n0 = tn * 16;

    for (int it = 0; it < ITER; ++it) {
        // --- M[j,a] = max_b sB[a,b]*X[j,b]  (branch-free; diag of sB is 0) ---
        for (int idx = tid; idx < NN * NN; idx += nt) {
            int j = idx / NN, a = idx % NN;
            const float* Br = sB + a * NN;
            const float* Xr = sX + j * NN;
            float m = -3.0e38f;
#pragma unroll 8
            for (int b = 0; b < NN; ++b) m = fmaxf(m, Br[b] * Xr[b]);
            sM[idx] = m;
        }
        __syncthreads();
        // --- Xn = X.*nsim + sA @ M  via f32 WMMA (wave-uniform branch) ---
        float ss = 0.0f;
        if (wave < 25) {
            v8f acc = {};
            // A frag (16x4 f32): lane -> row, VGPR v -> K = kb + 2*half + v
            // B frag (4x16 f32): lane -> col, VGPR v -> K row = kb + 2*half + v
            for (int kb = 0; kb < NN; kb += 4) {
                int k0 = kb + half * 2;
                v2f av, bv;
                av[0] = sA[(m0 + lidx) * NN + k0];
                av[1] = sA[(m0 + lidx) * NN + k0 + 1];
                bv[0] = sM[(k0) * NN + n0 + lidx];
                bv[1] = sM[(k0 + 1) * NN + n0 + lidx];
                acc = __builtin_amdgcn_wmma_f32_16x16x4_f32(
                    /*neg_a=*/false, av, /*neg_b=*/false, bv,
                    /*c_mod=*/(short)0, acc, /*reuse_a=*/false, /*reuse_b=*/false);
            }
#pragma unroll
            for (int r = 0; r < 8; ++r) {
                int i = m0 + half * 8 + r;
                int a = n0 + lidx;
                float nsim = 1.0f / (fabsf(sdA[i] - sdB[a]) + 1.0f);
                float v = sX[i * NN + a] * nsim + acc[r];
                sXn[i * NN + a] = v;
                ss += v * v;
            }
        }
        red[tid] = ss;
        __syncthreads();
        for (int s = nt >> 1; s > 0; s >>= 1) {
            if (tid < s) red[tid] += red[tid + s];
            __syncthreads();
        }
        float invn = 1.0f / sqrtf(red[0]);
        for (int idx = tid; idx < NN * NN; idx += nt) sX[idx] = sXn[idx] * invn;
        __syncthreads();
    }
    for (int idx = tid; idx < NN * NN; idx += nt) out[idx] = sX[idx];
}

// ---------------------------------------------------------------------------
extern "C" void kernel_launch(void* const* d_in, const int* in_sizes, int n_in,
                              void* d_out, int out_size, void* d_ws, size_t ws_size,
                              hipStream_t stream) {
    const float* x   = (const float*)d_in[0];
    const int*   ei  = (const int*)  d_in[1];
    const int*   adj = (const int*)  d_in[2];
    const float* eps = (const float*)d_in[3];
    const float* W1  = (const float*)d_in[4];
    const float* b1  = (const float*)d_in[5];
    const float* g1  = (const float*)d_in[6];
    const float* bt1 = (const float*)d_in[7];
    const float* W2  = (const float*)d_in[8];
    const float* b2  = (const float*)d_in[9];
    const float* g2  = (const float*)d_in[10];
    const float* bt2 = (const float*)d_in[11];
    const float* Wmu = (const float*)d_in[12];
    const float* bmu = (const float*)d_in[13];
    const float* Wlv = (const float*)d_in[14];
    const float* blv = (const float*)d_in[15];
    const float* We1 = (const float*)d_in[16];
    const float* be1 = (const float*)d_in[17];
    const float* We2 = (const float*)d_in[18];
    const float* be2 = (const float*)d_in[19];
    // d_in[20..23] (Wn1,bn1,Wn2,bn2) are dead in the reference output.

    float* ws   = (float*)d_ws;
    float* H1   = ws;                 // 80*256
    float* Hg1  = ws + 20480;         // 80*256
    float* H2   = ws + 40960;         // 80*256
    float* Hg2  = ws + 61440;         // 80*256
    float* dinv = ws + 81920;         // 80 (padded)
    float* he   = ws + 82048;         // 256
    float* vl   = ws + 82304;         // 3160

    k_deg<<<1, 128, 0, stream>>>(ei, dinv);

    // Layer 1: GEMM (80x64x256) -> aggregate -> BN+ReLU (in place)
    k_gemm_wmma<<<20, 128, 0, stream>>>(x, W1, H1, FF, HH);
    k_agg<<<NN, HH, 0, stream>>>(H1, dinv, ei, b1, Hg1);
    k_bn_relu<<<1, HH, 0, stream>>>(Hg1, g1, bt1);

    // Layer 2: GEMM (80x256x256) -> aggregate -> BN+ReLU (in place)
    k_gemm_wmma<<<20, 128, 0, stream>>>(Hg1, W2, H2, HH, HH);
    k_agg<<<NN, HH, 0, stream>>>(H2, dinv, ei, b2, Hg2);
    k_bn_relu<<<1, HH, 0, stream>>>(Hg2, g2, bt2);

    // Latent + decoder (fused), then edge logits
    k_latent_he<<<1, HH, 0, stream>>>(Hg2, Wmu, bmu, Wlv, blv, eps, We1, be1, he);
    k_vl<<<(OFF + HH - 1) / HH, HH, 0, stream>>>(he, We2, be2, vl);

    // MPM fixed point (builds A/B in LDS from adj + vec_logits)
    k_mpm<<<1, 1024, 0, stream>>>(adj, vl, (float*)d_out);
}
